// RCL_47699906789502
// MI455X (gfx1250) — matrix operand order repaired and checked by reading, not compiled, not run on hardware
//
#include <hip/hip_runtime.h>
#include <hip/hip_bf16.h>

typedef __attribute__((ext_vector_type(16))) _Float16 v16h;
typedef __attribute__((ext_vector_type(8)))  float    v8f;
typedef __attribute__((ext_vector_type(4)))  int      v4i;

#define ALPHA_C 0.001f
#define BETA_C  0.75f
#define HH 256
#define WW 256
#define BB 8
#define COUT 64

#if defined(__has_builtin)
#  if __has_builtin(__builtin_amdgcn_global_load_async_to_lds_b128)
#    define USE_ASYNC_LDS 1
#  endif
#endif
#ifndef USE_ASYNC_LDS
#  define USE_ASYNC_LDS 0
#endif

typedef __attribute__((address_space(1))) v4i* gv4i_p;   // global int4*
typedef __attribute__((address_space(3))) v4i* lv4i_p;   // LDS int4*

union F16Frag { v16h v; uint4 q[2]; _Float16 h[16]; };
union AccU   { v8f  v; float4 q[2]; float f[8]; };

// d^(-0.75) = rsqrt(d) * sqrt(rsqrt(d)); d >= 1 here so this is safe.
__device__ __forceinline__ float lrn_scale(float s) {
  float d = 1.f + ALPHA_C * (s * (1.f / 7.f));
  float r = __frsqrt_rn(d);
  return r * __fsqrt_rn(r);
}

__device__ __forceinline__ void copy16_to_lds(const _Float16* __restrict__ src,
                                              _Float16* dst) {
#if USE_ASYNC_LDS
  __builtin_amdgcn_global_load_async_to_lds_b128(
      (gv4i_p)(void*)src, (lv4i_p)(void*)dst, 0, 0);
#else
  *(uint4*)dst = *(const uint4*)src;
#endif
}

__device__ __forceinline__ void wait_async_lds() {
#if USE_ASYNC_LDS
#  if __has_builtin(__builtin_amdgcn_s_wait_asynccnt)
  __builtin_amdgcn_s_wait_asynccnt(0);
#  else
  asm volatile("s_wait_asynccnt 0" ::: "memory");
#  endif
#endif
}

// ---------------------------------------------------------------------------
// Pack OIHW f32 weights into per-lane WMMA A-fragment f16 layout:
// out[((tile*9+tap)*chunks+chunk)*512 + lane*16 + i]
// A 16x32 f16 layout (contiguous k): lane<16 -> m=lane,  k in {0..15}
//                                  lane>=16 -> m=lane-16, k in {16..31}
// ---------------------------------------------------------------------------
__global__ void pack_w_kernel(const float* __restrict__ w, _Float16* __restrict__ out, int cin) {
  int tid = blockIdx.x * blockDim.x + threadIdx.x;
  int chunks = cin >> 5;
  int total = 4 * 9 * chunks * 32;
  if (tid >= total) return;
  int lane = tid & 31;
  int t = tid >> 5;
  int chunk = t % chunks; t /= chunks;
  int tap = t % 9; t /= 9;
  int tile = t;
  int ky = tap / 3, kx = tap % 3;
  int m = lane & 15;
  int kbase = (lane < 16) ? 0 : 16;
  int cout = tile * 16 + m;
  _Float16* dst = out + (size_t)tid * 16;
#pragma unroll
  for (int i = 0; i < 16; ++i) {
    int k = kbase + i;
    int ci = chunk * 32 + k;
    dst[i] = (_Float16)w[((size_t)(cout * cin + ci)) * 9 + ky * 3 + kx];
  }
}

// ---------------------------------------------------------------------------
// x: NCHW f32 [8,32,256,256] -> NHWC f16 [8,256,256,32]
// ---------------------------------------------------------------------------
__global__ void nchw_to_nhwc16_kernel(const float* __restrict__ x, _Float16* __restrict__ xh) {
  int tid = blockIdx.x * blockDim.x + threadIdx.x;
  if (tid >= BB * HH * WW) return;
  int xc = tid & (WW - 1);
  int yr = (tid >> 8) & (HH - 1);
  int b  = tid >> 16;
  union { _Float16 h[32]; uint4 q[4]; } o;
#pragma unroll
  for (int c = 0; c < 32; ++c)
    o.h[c] = (_Float16)x[(((size_t)b * 32 + c) * HH + yr) * WW + xc];
  uint4* dst = (uint4*)(xh + (size_t)tid * 32);
#pragma unroll
  for (int q = 0; q < 4; ++q) dst[q] = o.q[q];
}

// ---------------------------------------------------------------------------
// Timestep 1 (rec input is zero): f = relu(u + rec_b + bias); LRN; -> f16 NHWC
// ---------------------------------------------------------------------------
__global__ void step1_kernel(const float* __restrict__ u, const float* __restrict__ rec_b,
                             const float* __restrict__ bias, _Float16* __restrict__ out) {
  int tid = blockIdx.x * blockDim.x + threadIdx.x;
  if (tid >= BB * HH * WW * 2) return;
  int chalf = tid & 1;
  int site = tid >> 1;
  int c0 = chalf * 32;
  const float* us = u + (size_t)site * 64;
  float fv[38], f2[38];
#pragma unroll
  for (int j = 0; j < 38; ++j) {
    int c = c0 + j - 3;
    float f = 0.f;
    if (c >= 0 && c < 64) f = fmaxf(us[c] + rec_b[c] + bias[c], 0.f);
    fv[j] = f; f2[j] = f * f;
  }
  float s = f2[0] + f2[1] + f2[2] + f2[3] + f2[4] + f2[5] + f2[6];
  union { _Float16 h[32]; uint4 q[4]; } o;
#pragma unroll
  for (int jj = 0; jj < 32; ++jj) {
    o.h[jj] = (_Float16)(fv[jj + 3] * lrn_scale(s));
    if (jj < 31) s += f2[jj + 7] - f2[jj];
  }
  uint4* dst = (uint4*)(out + (size_t)site * 64 + c0);
#pragma unroll
  for (int q = 0; q < 4; ++q) dst[q] = o.q[q];
}

// ---------------------------------------------------------------------------
// Implicit-GEMM 3x3 SAME conv via v_wmma_f32_16x16x32_f16.
// Block = 128 threads (4 waves); one (batch, row, 64-px strip); 64 couts.
// MODE 0: out_f = acc + bvec (ff conv -> u, NHWC f32)
// MODE 1: z = acc+u+bvec+bias2; relu; LRN; out_h f16 NHWC
// MODE 2: same but out_f f32 NCHW (final output)
// ---------------------------------------------------------------------------
template<int CH, int MODE>
__global__ __launch_bounds__(128)
void conv_wmma_kernel(const _Float16* __restrict__ in_h,
                      const _Float16* __restrict__ apack,
                      const float* __restrict__ u,
                      const float* __restrict__ bvec,
                      const float* __restrict__ bias2,
                      float* __restrict__ out_f,
                      _Float16* __restrict__ out_h) {
  constexpr int CHUNKS = CH / 32;
  constexpr int NFRAG  = 9 * CHUNKS;
  constexpr int PAD    = 8;                 // halfs; keeps b128 frag reads conflict-free
  constexpr int NV     = CH / 8;            // uint4s per pixel site
  __shared__ _Float16 tile[3][66][CH + PAD];
  __shared__ float fbuf[(MODE >= 1) ? 64 : 1][68];

  const int b  = blockIdx.z;
  const int y  = blockIdx.y;
  const int x0 = blockIdx.x * 64;
  const int tid  = threadIdx.x;
  const int lane = tid & 31;
  const int wave = tid >> 5;                // cout tile 0..3
  const int n    = lane & 15;               // N column within 16x16 tile
  const int kb   = (lane < 16) ? 0 : 16;    // B fragment K base for this lane
  const int moff = (lane < 16) ? 0 : 8;     // C/D fragment M base

  // Cooperative zero-padded halo load: rows y-1..y+1, x0-1..x0+64, NHWC f16.
  // In-range sites go straight to LDS via async DMA (ASYNCcnt); OOB sites get
  // plain LDS zero stores (DScnt, covered by the barrier).
  for (int s = tid; s < 3 * 66; s += 128) {
    int r = s / 66, xi = s - r * 66;
    int gy = y + r - 1, gx = x0 + xi - 1;
    _Float16* dstp = &tile[r][xi][0];
    if (gy >= 0 && gy < HH && gx >= 0 && gx < WW) {
      const _Float16* srcp = in_h + ((size_t)((b * HH + gy) * WW + gx)) * CH;
#pragma unroll
      for (int q = 0; q < NV; ++q) copy16_to_lds(srcp + q * 8, dstp + q * 8);
    } else {
      uint4 z; z.x = z.y = z.z = z.w = 0u;
      uint4* dq = (uint4*)dstp;
#pragma unroll
      for (int q = 0; q < NV; ++q) dq[q] = z;
    }
  }

  // Preload this wave's A fragments (weights) into registers: NFRAG x 32B.
  // Overlaps with the in-flight async tile fill.
  F16Frag afr[NFRAG];
  {
    const _Float16* ap = apack + (((size_t)wave * NFRAG) << 9) + ((size_t)lane << 4);
#pragma unroll
    for (int t = 0; t < NFRAG; ++t) {
      const uint4* p = (const uint4*)(ap + ((size_t)t << 9));
      afr[t].q[0] = p[0];
      afr[t].q[1] = p[1];
    }
  }

  const int cb = wave * 16 + moff;          // first of 8 couts this lane holds
  float bsum[8];
#pragma unroll
  for (int r = 0; r < 8; ++r)
    bsum[r] = bvec[cb + r] + ((MODE >= 1) ? bias2[cb + r] : 0.f);

  wait_async_lds();
  __syncthreads();

  const size_t site_base = ((size_t)(b * HH + y)) * WW + x0;

#pragma unroll
  for (int nt = 0; nt < 4; ++nt) {
    AccU acc;
#pragma unroll
    for (int r = 0; r < 8; ++r) acc.f[r] = 0.f;

#pragma unroll
    for (int t = 0; t < NFRAG; ++t) {
      const int tap = t / CHUNKS, chunk = t % CHUNKS;
      const int dy = tap / 3, dx = tap % 3;
      const int xi = nt * 16 + n + dx;      // 0..65 within halo tile
      F16Frag bf;                           // B 32x16: lane n=lane%16, k=kb+i
      const uint4* p = (const uint4*)&tile[dy][xi][chunk * 32 + kb];
      bf.q[0] = p[0];
      bf.q[1] = p[1];
      acc.v = __builtin_amdgcn_wmma_f32_16x16x32_f16(false, afr[t].v, false, bf.v,
                                                     (short)0, acc.v, false, false);
    }

    const int px = nt * 16 + n;
    const size_t site = site_base + px;
    if (MODE == 0) {
      float4 o0, o1;
      o0.x = acc.f[0] + bsum[0]; o0.y = acc.f[1] + bsum[1];
      o0.z = acc.f[2] + bsum[2]; o0.w = acc.f[3] + bsum[3];
      o1.x = acc.f[4] + bsum[4]; o1.y = acc.f[5] + bsum[5];
      o1.z = acc.f[6] + bsum[6]; o1.w = acc.f[7] + bsum[7];
      float4* dst = (float4*)(out_f + site * 64 + cb);
      dst[0] = o0; dst[1] = o1;
    } else {
      const float* up = u + site * 64 + cb;
      float4 u0 = *(const float4*)up;
      float4 u1 = *(const float4*)(up + 4);
      float4 z0, z1;
      z0.x = fmaxf(acc.f[0] + u0.x + bsum[0], 0.f);
      z0.y = fmaxf(acc.f[1] + u0.y + bsum[1], 0.f);
      z0.z = fmaxf(acc.f[2] + u0.z + bsum[2], 0.f);
      z0.w = fmaxf(acc.f[3] + u0.w + bsum[3], 0.f);
      z1.x = fmaxf(acc.f[4] + u1.x + bsum[4], 0.f);
      z1.y = fmaxf(acc.f[5] + u1.y + bsum[5], 0.f);
      z1.z = fmaxf(acc.f[6] + u1.z + bsum[6], 0.f);
      z1.w = fmaxf(acc.f[7] + u1.w + bsum[7], 0.f);
      float4* fd = (float4*)&fbuf[px][cb];
      fd[0] = z0; fd[1] = z1;
    }
  }

  if (MODE >= 1) {
    __syncthreads();
    // LRN over channels, zero-padded window of 7, always /7 (matches reference).
    const int px = tid & 63;
    const int chalf = tid >> 6;
    const int c0 = chalf * 32;
    float f2[38];
#pragma unroll
    for (int j = 0; j < 38; ++j) {
      int c = c0 + j - 3;
      float f = (c >= 0 && c < 64) ? fbuf[px][c] : 0.f;
      f2[j] = f * f;
    }
    float s = f2[0] + f2[1] + f2[2] + f2[3] + f2[4] + f2[5] + f2[6];
    const size_t site = site_base + px;
    if (MODE == 1) {
      union { _Float16 h[32]; uint4 q[4]; } o;
#pragma unroll
      for (int jj = 0; jj < 32; ++jj) {
        o.h[jj] = (_Float16)(fbuf[px][c0 + jj] * lrn_scale(s));
        if (jj < 31) s += f2[jj + 7] - f2[jj];
      }
      uint4* dst = (uint4*)(out_h + site * 64 + c0);
#pragma unroll
      for (int q = 0; q < 4; ++q) dst[q] = o.q[q];
    } else {
#pragma unroll
      for (int jj = 0; jj < 32; ++jj) {
        float val = fbuf[px][c0 + jj] * lrn_scale(s);
        out_f[(((size_t)b * COUT + (c0 + jj)) * HH + y) * WW + x0 + px] = val;
        if (jj < 31) s += f2[jj + 7] - f2[jj];
      }
    }
  }
}

// ---------------------------------------------------------------------------
extern "C" void kernel_launch(void* const* d_in, const int* in_sizes, int n_in,
                              void* d_out, int out_size, void* d_ws, size_t ws_size,
                              hipStream_t stream) {
  (void)in_sizes; (void)n_in; (void)out_size; (void)ws_size;
  const float* x     = (const float*)d_in[0];
  const float* ff_w  = (const float*)d_in[1];
  const float* ff_b  = (const float*)d_in[2];
  const float* rec_w = (const float*)d_in[3];
  const float* rec_b = (const float*)d_in[4];
  const float* bias  = (const float*)d_in[5];

  char* ws = (char*)d_ws;
  const size_t NPIX = (size_t)BB * HH * WW;                 // 524288 sites
  float*    u   = (float*)ws;                                // NHWC f32, 134MB
  _Float16* xA  = (_Float16*)(ws + NPIX * 64 * 4);           // NHWC f16, 67MB
  _Float16* xB  = (_Float16*)(ws + NPIX * 64 * 4 + NPIX * 64 * 2);
  _Float16* xh  = (_Float16*)(ws + NPIX * 64 * 4 + 2 * NPIX * 64 * 2);  // 33.5MB
  _Float16* ffp = (_Float16*)(ws + NPIX * 64 * 4 + 2 * NPIX * 64 * 2 + NPIX * 32 * 2);
  _Float16* rcp = ffp + (size_t)4 * 9 * 1 * 512;

  // 1) pack weights into WMMA A-fragment layout (f16)
  pack_w_kernel<<<dim3((1152 + 127) / 128), dim3(128), 0, stream>>>(ff_w, ffp, 32);
  pack_w_kernel<<<dim3((2304 + 127) / 128), dim3(128), 0, stream>>>(rec_w, rcp, 64);
  // 2) x NCHW f32 -> NHWC f16
  nchw_to_nhwc16_kernel<<<dim3((unsigned)(NPIX / 256)), dim3(256), 0, stream>>>(x, xh);
  // 3) u = conv(x, ff_w) + ff_b
  dim3 grid(WW / 64, HH, BB);
  conv_wmma_kernel<32, 0><<<grid, dim3(128), 0, stream>>>(xh, ffp, nullptr, ff_b, nullptr, u, nullptr);
  // 4) timestep 1 (rec conv of zeros): elementwise + LRN
  step1_kernel<<<dim3((unsigned)(NPIX * 2 / 256)), dim3(256), 0, stream>>>(u, rec_b, bias, xA);
  // 5) timestep 2: full rec conv + LRN -> f16 NHWC
  conv_wmma_kernel<64, 1><<<grid, dim3(128), 0, stream>>>(xA, rcp, u, rec_b, bias, nullptr, xB);
  // 6) timestep 3: full rec conv + LRN -> final f32 NCHW
  conv_wmma_kernel<64, 2><<<grid, dim3(128), 0, stream>>>(xB, rcp, u, rec_b, bias, (float*)d_out, nullptr);
}
